// LlamaAttentionGIN_19164144074841
// MI455X (gfx1250) — matrix operand order repaired and checked by reading, not compiled, not run on hardware
//
#include <hip/hip_runtime.h>
#include <hip/hip_bf16.h>

// ---------------- problem constants ----------------
constexpr int B_ = 2, S_ = 1024, HID_ = 1024, H_ = 16, D_ = 64, E_ = 128;
constexpr float THRESH = 0.2f;
constexpr float RMS_EPS = 1e-6f;

typedef __attribute__((ext_vector_type(16))) __bf16 v16bf;
typedef __attribute__((ext_vector_type(8)))  float  v8f;

// ---------------- bf16 helpers ----------------
__device__ __forceinline__ unsigned short f2bfbits(float f) {
    unsigned u = __builtin_bit_cast(unsigned, f);
    unsigned r = u + 0x7FFFu + ((u >> 16) & 1u);   // round-to-nearest-even
    return (unsigned short)(r >> 16);
}
__device__ __forceinline__ __bf16 usbf(unsigned short u) {
    return __builtin_bit_cast(__bf16, u);
}
// load 8 contiguous bf16 (16B aligned) into a[off..off+8)
__device__ __forceinline__ void load8(const unsigned short* p, v16bf& a, int off) {
    uint4 w = *reinterpret_cast<const uint4*>(p);
    a[off + 0] = usbf((unsigned short)(w.x));       a[off + 1] = usbf((unsigned short)(w.x >> 16));
    a[off + 2] = usbf((unsigned short)(w.y));       a[off + 3] = usbf((unsigned short)(w.y >> 16));
    a[off + 4] = usbf((unsigned short)(w.z));       a[off + 5] = usbf((unsigned short)(w.z >> 16));
    a[off + 6] = usbf((unsigned short)(w.w));       a[off + 7] = usbf((unsigned short)(w.w >> 16));
}

// A-tile (16x32 bf16) loader. src points at tile origin (row-major, leading dim ld).
// ISA 7.12.2 16-bit A layout: lane<16 holds row r = lane, K = {0..7, 16..23};
// lane>=16 holds row lane-16, K = {8..15, 24..31}.
__device__ __forceinline__ v16bf load_a(const unsigned short* src, int ld, int lane) {
    int r = lane & 15, half = lane >> 4;
    const unsigned short* p = src + (size_t)r * ld + half * 8;
    v16bf a;
    load8(p,      a, 0);     // K = half*8 .. +8
    load8(p + 16, a, 8);     // K = 16 + half*8 .. +8
    return a;
}
// B-tile (32x16 bf16) loader from Bt = B^T row-major [N][K] at tile origin.
// lane holds column c = lane&15; lanes<16: K={0..7,16..23}, lanes>=16: K={8..15,24..31}.
__device__ __forceinline__ v16bf load_b(const unsigned short* bt, int ld, int lane) {
    int c = lane & 15, half = lane >> 4;
    const unsigned short* p = bt + (size_t)c * ld + half * 8;
    v16bf b;
    load8(p,     b, 0);
    load8(p + 16, b, 8);
    return b;
}
__device__ __forceinline__ v8f wmma_bf16(v16bf a, v16bf b, v8f c) {
    return __builtin_amdgcn_wmma_f32_16x16x32_bf16(false, a, false, b, (short)0, c, false, false);
}

__device__ __forceinline__ int wred_add(int v) {
#pragma unroll
    for (int o = 16; o; o >>= 1) v += __shfl_xor(v, o, 32);
    return v;
}

// ---------------- kernel 0: fp32 -> bf16 convert / transpose-convert ----------------
__global__ void conv_bf(const float* __restrict__ src, unsigned short* __restrict__ dst, int n) {
    int i = blockIdx.x * 256 + threadIdx.x;
    if (i < n) dst[i] = f2bfbits(src[i]);
}
// src[h][r][c] (rows x cols) -> dst[h][c][r]
__global__ void convT_bf(const float* __restrict__ src, unsigned short* __restrict__ dst,
                         int rows, int cols) {
    int h = blockIdx.y;
    int n = rows * cols;
    int i = blockIdx.x * 256 + threadIdx.x;
    if (i < n) {
        int r = i / cols, c = i % cols;
        dst[(size_t)h * n + (size_t)c * rows + r] = f2bfbits(src[(size_t)h * n + i]);
    }
}

// ---------------- kernel 1: Q/K projection (WMMA) + RoPE ----------------
// grid.x = (B*S/16)*H, grid.y = 2 (0:q 1:k), block = 128 (4 waves; wave w does N-subtile w)
__global__ __launch_bounds__(128) void qk_proj_rope(
    const unsigned short* __restrict__ hid_bf,
    const unsigned short* __restrict__ wq_bf,
    const unsigned short* __restrict__ wk_bf,
    unsigned short* __restrict__ q_bf,
    unsigned short* __restrict__ k_bf)
{
    __shared__ float tile[16 * 64];
    const int which = blockIdx.y;
    const int h     = blockIdx.x % H_;
    const int mtile = blockIdx.x / H_;
    const int wid = threadIdx.x >> 5, lane = threadIdx.x & 31;
    const int mbase = mtile * 16;
    const int ncol  = h * 64 + wid * 16;
    const unsigned short* aSrc = hid_bf + (size_t)mbase * HID_;
    const unsigned short* bt   = (which ? wk_bf : wq_bf) + (size_t)ncol * HID_;

    v8f acc = {};
    for (int k0 = 0; k0 < HID_; k0 += 32) {
        v16bf a = load_a(aSrc + k0, HID_, lane);
        v16bf b = load_b(bt + k0,   HID_, lane);
        acc = wmma_bf16(a, b, acc);
    }
    int col = lane & 15, half = lane >> 4;
#pragma unroll
    for (int v = 0; v < 8; ++v)
        tile[(v + 8 * half) * 64 + wid * 16 + col] = acc[v];
    __syncthreads();

    unsigned short* dst = which ? k_bf : q_bf;
    for (int e = threadIdx.x; e < 16 * 64; e += 128) {
        int r = e >> 6, d = e & 63;
        int m = mbase + r;
        int b_ = m / S_, s = m % S_;
        int i = d & 31;
        float inv = expf(-0.28782313662425572f * (float)i);  // ln(10000)/32
        float ang = (float)s * inv;
        float cs = cosf(ang), sn = sinf(ang);
        float x     = tile[r * 64 + d];
        float other = tile[r * 64 + (d < 32 ? d + 32 : d - 32)];
        float rot   = (d < 32) ? -other : other;
        float val = x * cs + rot * sn;
        dst[(((size_t)b_ * H_ + h) * S_ + s) * 64 + d] = f2bfbits(val);
    }
}

// ---------------- kernel 2: scores (WMMA) + exact threshold/top-k + neighbor-sum + GIN-in ----
// grid = (S/16, H, B), block = 128, dynamic LDS = 16*S*4 bytes
__global__ __launch_bounds__(128) void score_topk_gin(
    const unsigned short* __restrict__ q_bf,
    const unsigned short* __restrict__ k_bf,
    const float* __restrict__ hidden,
    const float* __restrict__ gin_eps,
    unsigned short* __restrict__ gin_bf)
{
    extern __shared__ float sc[];   // [16][S]
    const int t = blockIdx.x, h = blockIdx.y, b = blockIdx.z;
    const int wid = threadIdx.x >> 5, lane = threadIdx.x & 31;
    const int qbase = t * 16;

    // Phase A: score stripe = (Q[16x64] * K^T) / 8 for col tiles 0..t
    const unsigned short* aSrc = q_bf + (((size_t)b * H_ + h) * S_ + qbase) * 64;
    v16bf a0 = load_a(aSrc,      64, lane);
    v16bf a1 = load_a(aSrc + 32, 64, lane);
    const int col = lane & 15, half = lane >> 4;
    for (int jt = wid; jt <= t; jt += 4) {
        const unsigned short* bSrc = k_bf + (((size_t)b * H_ + h) * S_ + (size_t)jt * 16) * 64;
        v16bf b0 = load_b(bSrc,      64, lane);
        v16bf b1 = load_b(bSrc + 32, 64, lane);
        v8f acc = {};
        acc = wmma_bf16(a0, b0, acc);
        acc = wmma_bf16(a1, b1, acc);
#pragma unroll
        for (int v = 0; v < 8; ++v)
            sc[(v + 8 * half) * S_ + jt * 16 + col] = acc[v] * 0.125f;
    }
    __syncthreads();

    // Phase B: per-row exact selection + gather (wave w handles rows w, w+4, w+8, w+12)
    const float eps = gin_eps[h];
    const unsigned TB = __builtin_bit_cast(unsigned, 0.2f);
    for (int rr = 0; rr < 4; ++rr) {
        const int r = wid + rr * 4;
        const int qrow = qbase + r;
        const int n = qrow;                 // strict causal: cols 0..qrow-1
        const float* row = sc + (size_t)r * S_;
        float2 acc = make_float2(0.f, 0.f);
        if (n > 0) {
            int kk = (int)ceilf(0.1f * (float)qrow);
            if (kk < 1) kk = 1;
            auto cge = [&](unsigned x) -> int {   // count of filtered values with bits >= x
                int c = 0;
                for (int j = lane; j < n; j += 32) {
                    float f = row[j];
                    if (f >= THRESH) c += (__builtin_bit_cast(unsigned, f) >= x);
                }
                return wred_add(c);
            };
            int cnt = cge(TB);
            unsigned vk; int quota;
            if (cnt >= kk) {
                unsigned lo = TB, hi = 0x7F800000u;   // exact select on IEEE bits (monotone, >=0)
                while (lo < hi) {
                    unsigned mid = lo + ((hi - lo + 1) >> 1);
                    if (cge(mid) >= kk) lo = mid; else hi = mid - 1;
                }
                vk = lo;
                quota = kk - cge(vk + 1);             // ties at vk, lowest index first
            } else {
                vk = 0u;                              // all >=thresh kept + zeros by index
                quota = kk - cnt;
            }
            int eqt = 0;
            for (int j = 0; j < n; ++j) {
                float f = row[j];
                f = (f >= THRESH) ? f : 0.f;          // filtered = s*(s>=0.2)
                unsigned fb = __builtin_bit_cast(unsigned, f);
                bool keep = false;
                if (fb > vk) keep = true;
                else if (fb == vk && eqt < quota) { keep = true; ++eqt; }
                if (keep) {
                    float2 x = *(reinterpret_cast<const float2*>(
                        hidden + ((size_t)(b * S_ + j)) * HID_ + h * 64) + lane);
                    acc.x += x.x; acc.y += x.y;
                }
            }
        }
        float2 xq = *(reinterpret_cast<const float2*>(
            hidden + ((size_t)(b * S_ + qrow)) * HID_ + h * 64) + lane);
        float g0 = eps * xq.x + acc.x;
        float g1 = eps * xq.y + acc.y;
        size_t o = (((size_t)b * H_ + h) * S_ + qrow) * 64 + 2 * lane;
        gin_bf[o]     = f2bfbits(g0);
        gin_bf[o + 1] = f2bfbits(g1);
    }
}

// ---------------- kernel 3: h1 = gin @ W1 + b1, RMSNorm, SiLU (WMMA) ----------------
// grid = (S/16, H, B), block = 256 (8 waves; wave w = N-subtile w of E=128)
__global__ __launch_bounds__(256) void gin_mlp1(
    const unsigned short* __restrict__ gin_bf,
    const unsigned short* __restrict__ w1t_bf,   // [H][E][D]
    const float* __restrict__ b1,                // [H][E]
    const float* __restrict__ norm_w,            // [H][E]
    unsigned short* __restrict__ h1_bf)          // [B,H,S,E]
{
    __shared__ float hbuf[16 * 128];
    __shared__ float part[16 * 16];
    __shared__ float rmsv[16];
    const int t = blockIdx.x, h = blockIdx.y, b = blockIdx.z;
    const int wid = threadIdx.x >> 5, lane = threadIdx.x & 31;

    const unsigned short* aSrc = gin_bf + (((size_t)b * H_ + h) * S_ + (size_t)t * 16) * 64;
    const unsigned short* bt   = w1t_bf + (size_t)h * E_ * D_ + (size_t)(wid * 16) * D_;
    v16bf a0 = load_a(aSrc,      64, lane);
    v16bf a1 = load_a(aSrc + 32, 64, lane);
    v16bf b0 = load_b(bt,        64, lane);
    v16bf b1v = load_b(bt + 32,  64, lane);
    v8f acc = {};
    acc = wmma_bf16(a0, b0, acc);
    acc = wmma_bf16(a1, b1v, acc);

    const int col = lane & 15, half = lane >> 4;
    const float bias = b1[h * E_ + wid * 16 + col];
#pragma unroll
    for (int v = 0; v < 8; ++v)
        hbuf[(v + 8 * half) * 128 + wid * 16 + col] = acc[v] + bias;
    __syncthreads();

    const int r = threadIdx.x >> 4, c16 = threadIdx.x & 15;
    float s = 0.f;
#pragma unroll
    for (int i = 0; i < 8; ++i) { float x = hbuf[r * 128 + c16 * 8 + i]; s += x * x; }
    part[r * 16 + c16] = s;
    __syncthreads();
    if (threadIdx.x < 16) {
        float ss = 0.f;
#pragma unroll
        for (int i = 0; i < 16; ++i) ss += part[threadIdx.x * 16 + i];
        rmsv[threadIdx.x] = rsqrtf(ss * (1.f / 128.f) + RMS_EPS);
    }
    __syncthreads();
    const float rm = rmsv[r];
    size_t orow = (((size_t)b * H_ + h) * S_ + (size_t)t * 16 + r) * E_;
#pragma unroll
    for (int i = 0; i < 8; ++i) {
        int e = c16 * 8 + i;
        float v = hbuf[r * 128 + e] * rm * norm_w[h * E_ + e];
        v = v / (1.f + expf(-v));     // SiLU
        h1_bf[orow + e] = f2bfbits(v);
    }
}

// ---------------- kernel 4: h2 = h1 @ W2 + b2, transpose back to [B,S,HID] bf16 ----------------
// grid = (S/16, H, B), block = 128 (4 waves; wave w = N-subtile w of D=64)
__global__ __launch_bounds__(128) void mlp2_attn(
    const unsigned short* __restrict__ h1_bf,
    const unsigned short* __restrict__ w2t_bf,   // [H][D][E]
    const float* __restrict__ b2,                // [H][D]
    unsigned short* __restrict__ attn_bf)        // [B,S,HID]
{
    const int t = blockIdx.x, h = blockIdx.y, b = blockIdx.z;
    const int wid = threadIdx.x >> 5, lane = threadIdx.x & 31;
    const unsigned short* aSrc = h1_bf + (((size_t)b * H_ + h) * S_ + (size_t)t * 16) * E_;
    const unsigned short* bt   = w2t_bf + (size_t)h * D_ * E_ + (size_t)(wid * 16) * E_;
    v8f acc = {};
#pragma unroll
    for (int k0 = 0; k0 < E_; k0 += 32) {
        v16bf a = load_a(aSrc + k0, E_, lane);
        v16bf bb = load_b(bt + k0,  E_, lane);
        acc = wmma_bf16(a, bb, acc);
    }
    const int col = lane & 15, half = lane >> 4;
    const float bias = b2[h * D_ + wid * 16 + col];
#pragma unroll
    for (int v = 0; v < 8; ++v) {
        int s = t * 16 + v + 8 * half;
        attn_bf[((size_t)b * S_ + s) * HID_ + h * 64 + wid * 16 + col] = f2bfbits(acc[v] + bias);
    }
}

// ---------------- kernel 5: out = attn @ Wo^T (WMMA, fp32 out) ----------------
// 8192 wave-tiles (128 M-tiles x 64 N-tiles), block = 256 (8 waves)
__global__ __launch_bounds__(256) void out_proj(
    const unsigned short* __restrict__ attn_bf,
    const unsigned short* __restrict__ wo_bf,    // [N][K] row-major = B^T
    float* __restrict__ out)
{
    const int w = blockIdx.x * 8 + (threadIdx.x >> 5);
    const int lane = threadIdx.x & 31;
    const int mt = w >> 6, nt = w & 63;
    const unsigned short* aSrc = attn_bf + (size_t)(mt * 16) * HID_;
    const unsigned short* bt   = wo_bf   + (size_t)(nt * 16) * HID_;
    v8f acc = {};
    for (int k0 = 0; k0 < HID_; k0 += 32) {
        v16bf a = load_a(aSrc + k0, HID_, lane);
        v16bf b = load_b(bt + k0,   HID_, lane);
        acc = wmma_bf16(a, b, acc);
    }
    const int col = lane & 15, half = lane >> 4;
#pragma unroll
    for (int v = 0; v < 8; ++v)
        out[(size_t)(mt * 16 + v + 8 * half) * HID_ + nt * 16 + col] = acc[v];
}

// ---------------- host launcher ----------------
extern "C" void kernel_launch(void* const* d_in, const int* in_sizes, int n_in,
                              void* d_out, int out_size, void* d_ws, size_t ws_size,
                              hipStream_t stream) {
    const float* hidden  = (const float*)d_in[0];
    const float* Wq      = (const float*)d_in[1];
    const float* Wk      = (const float*)d_in[2];
    const float* Wo      = (const float*)d_in[3];
    const float* gin_eps = (const float*)d_in[4];
    const float* W1      = (const float*)d_in[5];
    const float* b1      = (const float*)d_in[6];
    const float* norm_w  = (const float*)d_in[7];
    const float* W2      = (const float*)d_in[8];
    const float* b2      = (const float*)d_in[9];

    char* ws = (char*)d_ws;
    size_t off = 0;
    auto alloc = [&](size_t bytes) -> void* {
        void* p = ws + off;
        off += (bytes + 255) & ~(size_t)255;
        return p;
    };
    unsigned short* hid_bf = (unsigned short*)alloc((size_t)2 * B_ * S_ * HID_);
    unsigned short* wq_bf  = (unsigned short*)alloc((size_t)2 * HID_ * HID_);
    unsigned short* wk_bf  = (unsigned short*)alloc((size_t)2 * HID_ * HID_);
    unsigned short* wo_bf  = (unsigned short*)alloc((size_t)2 * HID_ * HID_);
    unsigned short* w1t_bf = (unsigned short*)alloc((size_t)2 * H_ * D_ * E_);
    unsigned short* w2t_bf = (unsigned short*)alloc((size_t)2 * H_ * E_ * D_);
    unsigned short* q_bf   = (unsigned short*)alloc((size_t)2 * B_ * H_ * S_ * D_);
    unsigned short* k_bf   = (unsigned short*)alloc((size_t)2 * B_ * H_ * S_ * D_);
    unsigned short* gin_bf = (unsigned short*)alloc((size_t)2 * B_ * H_ * S_ * D_);
    unsigned short* h1_bf  = (unsigned short*)alloc((size_t)2 * B_ * H_ * S_ * E_);
    unsigned short* attn_bf= (unsigned short*)alloc((size_t)2 * B_ * S_ * HID_);

    const int nHid = B_ * S_ * HID_;
    const int nW   = HID_ * HID_;
    conv_bf<<<(nHid + 255) / 256, 256, 0, stream>>>(hidden, hid_bf, nHid);
    conv_bf<<<(nW + 255) / 256, 256, 0, stream>>>(Wq, wq_bf, nW);
    conv_bf<<<(nW + 255) / 256, 256, 0, stream>>>(Wk, wk_bf, nW);
    conv_bf<<<(nW + 255) / 256, 256, 0, stream>>>(Wo, wo_bf, nW);
    convT_bf<<<dim3((D_ * E_ + 255) / 256, H_), 256, 0, stream>>>(W1, w1t_bf, D_, E_);
    convT_bf<<<dim3((E_ * D_ + 255) / 256, H_), 256, 0, stream>>>(W2, w2t_bf, E_, D_);

    qk_proj_rope<<<dim3((B_ * S_ / 16) * H_, 2), 128, 0, stream>>>(
        hid_bf, wq_bf, wk_bf, q_bf, k_bf);

    score_topk_gin<<<dim3(S_ / 16, H_, B_), 128, (size_t)16 * S_ * 4, stream>>>(
        q_bf, k_bf, hidden, gin_eps, gin_bf);

    gin_mlp1<<<dim3(S_ / 16, H_, B_), 256, 0, stream>>>(
        gin_bf, w1t_bf, b1, norm_w, h1_bf);

    mlp2_attn<<<dim3(S_ / 16, H_, B_), 128, 0, stream>>>(
        h1_bf, w2t_bf, b2, attn_bf);

    out_proj<<<(128 * 64) / 8, 256, 0, stream>>>(attn_bf, wo_bf, (float*)d_out);
}
